// VINN_12292196401695
// MI455X (gfx1250) — matrix operand order repaired and checked by reading, not compiled, not run on hardware
//
#include <hip/hip_runtime.h>
#include <hip/hip_bf16.h>

// ---------------------------------------------------------------------------
// VINN: encode -> cdist -> top-5 -> softmax-weighted action average
// MI455X plan: single fused streaming pass over reps (204.8MB, HBM-bound
// ~8.8us @ 23.3TB/s). Distance GEMM via v_wmma_f32_16x16x32_bf16 so matmul
// throughput is never the limiter. Top-k runs on SQUARED distances (sqrt is
// monotonic); sqrt happens only for the final 5 per row in the tiny merge
// kernel. Block-local top-5 in registers, candidate merge at the end.
// ---------------------------------------------------------------------------

typedef __attribute__((ext_vector_type(16))) __bf16   v16bf;
typedef __attribute__((ext_vector_type(8)))  float    v8f;
typedef __attribute__((ext_vector_type(4)))  float    vf4;
typedef __attribute__((ext_vector_type(4)))  unsigned vu4;

#define N_REPS 400000
#define D_FEAT 128
#define B_ROWS 32
#define A_DIM  7
#define K_TOP  5

#define NBLK  392          // workgroups in distance pass
#define ITERS 8            // 392*8*128 = 401408 >= 400000
#define COLS  128          // reps per block-iteration (8 waves x 16)
#define SBU   68           // uint stride of one rep row in LDS (128 bf16 + pad), 16B aligned
#define SD    132          // float stride of one dist row in LDS

union ABReg { v16bf v; unsigned u[8]; vu4 q[2]; };

__device__ __forceinline__ unsigned pack_bf16x2(float a, float b) {
  unsigned short ua = __builtin_bit_cast(unsigned short, (__bf16)a);
  unsigned short ub = __builtin_bit_cast(unsigned short, (__bf16)b);
  return (unsigned)ua | ((unsigned)ub << 16);
}

// sorted-ascending top-5 insert (register arrays, fully unrolled)
__device__ __forceinline__ void insert5(float d, int idx, float (&d5)[5], int (&i5)[5]) {
  if (d < d5[4]) {
    d5[4] = d; i5[4] = idx;
#pragma unroll
    for (int q = 4; q > 0; --q) {
      if (d5[q] < d5[q-1]) {
        float td = d5[q]; d5[q] = d5[q-1]; d5[q-1] = td;
        int   ti = i5[q]; i5[q] = i5[q-1]; i5[q-1] = ti;
      }
    }
  }
}

// ---------------------------------------------------------------------------
// Kernel 1: batch_rep = x @ W_enc + b_enc  (f32, exact), plus row norms.
// grid 32 (one block per batch row), 256 threads.
// ---------------------------------------------------------------------------
__global__ void __launch_bounds__(256) enc_kernel(
    const float* __restrict__ x, const float* __restrict__ W,
    const float* __restrict__ b, float* __restrict__ brG, float* __restrict__ brnG)
{
  __shared__ float part[256];
  const int m = blockIdx.x;
  const int t = threadIdx.x;
  const int d = t & 127, h = t >> 7;   // two threads per output column

  const float* xr = x + (size_t)m * 2048 + h * 1024;
  const float* wp = W + (size_t)h * 1024 * 128 + d;
  float acc = 0.f;
#pragma unroll 8
  for (int j = 0; j < 1024; ++j) acc += xr[j] * wp[(size_t)j * 128];

  part[t] = acc;
  __syncthreads();
  float br = 0.f;
  if (t < 128) br = part[t] + part[t + 128] + b[t];
  __syncthreads();
  part[t] = (t < 128) ? br * br : 0.f;
  if (t < 128) brG[m * 128 + t] = br;
  __syncthreads();
#pragma unroll
  for (int off = 128; off >= 1; off >>= 1) {
    if (t < off) part[t] += part[t + off];
    __syncthreads();
  }
  if (t == 0) brnG[m] = part[0];
}

// ---------------------------------------------------------------------------
// Kernel 2: fused distance GEMM (WMMA bf16) + streaming per-block top-5
// over SQUARED distances (no sqrt in the hot loop).
// grid NBLK, 256 threads = 8 waves. Each wave owns a 16-rep column tile.
// ---------------------------------------------------------------------------
__global__ void __launch_bounds__(256) dist_topk_kernel(
    const float* __restrict__ reps, const float* __restrict__ brG,
    const float* __restrict__ brnG, float* __restrict__ candD, int* __restrict__ candI)
{
  __shared__ __align__(16) unsigned ldsB[128 * SBU];   // staged reps, bf16 pairs
  __shared__ float ldsNorm[128];                        // staged rep norms (f32)
  __shared__ float ldsDist[32 * SD];                    // 32 x 128 sq-dist tile
  __shared__ float ldsCD[32 * 40];                      // per-row merge buffer
  __shared__ int   ldsCI[32 * 40];

  const int t    = threadIdx.x;
  const int lane = t & 31, w = t >> 5;
  const int ll   = lane & 15, lh = lane >> 4;

  // --- A matrix (batch_rep) in exact WMMA bf16 A-layout, resident in VGPRs ---
  // A element for VGPR v, lane: M = lane&15, K = 32*kc + 2*(v&3) + 8*(lane>>4) + 16*(v>>2) (+half)
  ABReg a[2][4];
#pragma unroll
  for (int mt = 0; mt < 2; ++mt)
#pragma unroll
    for (int kc = 0; kc < 4; ++kc)
#pragma unroll
      for (int v = 0; v < 8; ++v) {
        const int K0 = 32 * kc + 2 * (v & 3) + 8 * lh + 16 * (v >> 2);
        const float* p = brG + (size_t)(16 * mt + ll) * 128 + K0;
        a[mt][kc].u[v] = pack_bf16x2(p[0], p[1]);
      }
  float brn0[8], brn1[8];
#pragma unroll
  for (int c = 0; c < 8; ++c) {
    brn0[c] = brnG[c + 8 * lh];
    brn1[c] = brnG[16 + c + 8 * lh];
  }

  float d5[5]; int i5[5];
#pragma unroll
  for (int q = 0; q < 5; ++q) { d5[q] = 1e30f; i5[q] = 0; }

  const int r_scan = t >> 3;     // scan-phase row (0..31)
  const int s_scan = t & 7;      // scan-phase segment (0..7)

  for (int it = 0; it < ITERS; ++it) {
    const int base = (blockIdx.x * ITERS + it) * COLS;

    // --- stage 128 reps -> LDS as packed bf16, nontemporal (reps > L2) ---
    // wave w handles rep (w + 8*i); lane handles features lane*4..lane*4+3
    for (int i = 0; i < 16; ++i) {
      const int r = w + 8 * i;
      const int n = base + r;
      vf4 f = {0.f, 0.f, 0.f, 0.f};
      if (n < N_REPS)
        f = __builtin_nontemporal_load(((const vf4*)(reps + (size_t)n * D_FEAT)) + lane);
      ldsB[r * SBU + lane * 2]     = pack_bf16x2(f.x, f.y);
      ldsB[r * SBU + lane * 2 + 1] = pack_bf16x2(f.z, f.w);
      float nrm = f.x * f.x + f.y * f.y + f.z * f.z + f.w * f.w;
#pragma unroll
      for (int off = 16; off > 0; off >>= 1) nrm += __shfl_xor(nrm, off, 32);
      if (lane == 0) ldsNorm[r] = (n < N_REPS) ? nrm : 1e30f;
    }
    __syncthreads();

    // --- GEMM: wave w -> cols [w*16, w*16+16), rows 0..31, K=128 in 4 chunks ---
    v8f acc0 = {}; v8f acc1 = {};
    const int repL = w * 16 + ll;
#pragma unroll
    for (int kc = 0; kc < 4; ++kc) {
      ABReg bb;
      const vu4* bp = (const vu4*)(ldsB + repL * SBU + kc * 16 + 4 * lh);
      bb.q[0] = bp[0];                       // ds_load_b128
      bb.q[1] = bp[2];
      acc0 = __builtin_amdgcn_wmma_f32_16x16x32_bf16(
          false, a[0][kc].v, false, bb.v, (short)0, acc0, false, false);
      acc1 = __builtin_amdgcn_wmma_f32_16x16x32_bf16(
          false, a[1][kc].v, false, bb.v, (short)0, acc1, false, false);
    }

    // --- dot -> SQUARED distance, scatter into LDS tile (sqrt deferred) ---
    const float rn  = ldsNorm[repL];
    const int  colL = w * 16 + ll;
#pragma unroll
    for (int c = 0; c < 8; ++c) {
      float sq0 = brn0[c] + rn - 2.f * acc0[c];
      float sq1 = brn1[c] + rn - 2.f * acc1[c];
      ldsDist[(c + 8 * lh) * SD + colL]      = sq0;
      ldsDist[(16 + c + 8 * lh) * SD + colL] = sq1;
    }
    __syncthreads();

    // --- scan: thread (row, seg) folds 16 sq-distances into its top-5 ---
    if (base + COLS <= N_REPS) {
      // fast path: whole tile in range (all tiles except the last 11, in blocks 390..391)
#pragma unroll
      for (int j = 0; j < 16; ++j) {
        const int col = s_scan * 16 + j;
        insert5(ldsDist[r_scan * SD + col], base + col, d5, i5);
      }
    } else {
#pragma unroll
      for (int j = 0; j < 16; ++j) {
        const int col = s_scan * 16 + j;
        const int idx = base + col;
        if (idx < N_REPS) insert5(ldsDist[r_scan * SD + col], idx, d5, i5);
      }
    }
    __syncthreads();
  }

  // --- merge 8x top-5 per row -> block top-5 per row, write candidates ---
#pragma unroll
  for (int q = 0; q < 5; ++q) {
    ldsCD[r_scan * 40 + s_scan * 5 + q] = d5[q];
    ldsCI[r_scan * 40 + s_scan * 5 + q] = i5[q];
  }
  __syncthreads();
  if (s_scan == 0) {
    for (int p = 0; p < K_TOP; ++p) {
      float mb = 1e30f; int mi = 0, mj = 0;
      for (int j = 0; j < 40; ++j) {
        float v = ldsCD[r_scan * 40 + j];
        if (v < mb) { mb = v; mi = ldsCI[r_scan * 40 + j]; mj = j; }
      }
      ldsCD[r_scan * 40 + mj] = 1e30f;
      candD[((size_t)r_scan * NBLK + blockIdx.x) * K_TOP + p] = mb;
      candI[((size_t)r_scan * NBLK + blockIdx.x) * K_TOP + p] = mi;
    }
  }
}

// ---------------------------------------------------------------------------
// Kernel 3: merge NBLK*5 sq-dist candidates/row -> global top-5, then
// dist = sqrt(max(sq,1e-12)), softmax(-dist), gather actions, weighted avg.
// 1 block, 256 threads (8 per row).
// ---------------------------------------------------------------------------
__global__ void __launch_bounds__(256) finalize_kernel(
    const float* __restrict__ candD, const int* __restrict__ candI,
    const float* __restrict__ actions, float* __restrict__ out)
{
  __shared__ float ldsCD[32 * 40];
  __shared__ int   ldsCI[32 * 40];
  const int t = threadIdx.x;
  const int r = t >> 3, s = t & 7;
  const int M = NBLK * K_TOP;

  float d5[5]; int i5[5];
#pragma unroll
  for (int q = 0; q < 5; ++q) { d5[q] = 1e30f; i5[q] = 0; }

  for (int j = s; j < M; j += 8) {
    float dd = candD[(size_t)r * M + j];
    int  idx = candI[(size_t)r * M + j];
    insert5(dd, idx, d5, i5);
  }
#pragma unroll
  for (int q = 0; q < 5; ++q) {
    ldsCD[r * 40 + s * 5 + q] = d5[q];
    ldsCI[r * 40 + s * 5 + q] = i5[q];
  }
  __syncthreads();

  if (s == 0) {
    float bd[5]; int bi[5];
    for (int p = 0; p < K_TOP; ++p) {
      float mb = 1e30f; int mi = 0, mj = 0;
      for (int j = 0; j < 40; ++j) {
        float v = ldsCD[r * 40 + j];
        if (v < mb) { mb = v; mi = ldsCI[r * 40 + j]; mj = j; }
      }
      ldsCD[r * 40 + mj] = 1e30f;
      // only here do we pay for sqrt (5 values per row)
      bd[p] = __builtin_sqrtf(fmaxf(mb, 1e-12f));
      bi[p] = mi;
    }
    // softmax(-d), stabilized with d_min (= bd[0], ascending order)
    const float dmin = bd[0];
    float e[5], sum = 0.f;
#pragma unroll
    for (int q = 0; q < 5; ++q) { e[q] = __expf(dmin - bd[q]); sum += e[q]; }
    const float inv = 1.f / sum;
#pragma unroll
    for (int a = 0; a < A_DIM; ++a) {
      float o = 0.f;
#pragma unroll
      for (int q = 0; q < 5; ++q)
        o += e[q] * inv * actions[(size_t)bi[q] * A_DIM + a];
      out[r * A_DIM + a] = o;
    }
  }
}

// ---------------------------------------------------------------------------
extern "C" void kernel_launch(void* const* d_in, const int* in_sizes, int n_in,
                              void* d_out, int out_size, void* d_ws, size_t ws_size,
                              hipStream_t stream) {
  const float* x       = (const float*)d_in[0];   // [32, 2048]
  const float* W_enc   = (const float*)d_in[1];   // [2048, 128]
  const float* b_enc   = (const float*)d_in[2];   // [128]
  const float* reps    = (const float*)d_in[3];   // [400000, 128]
  const float* actions = (const float*)d_in[4];   // [400000, 7]
  float* out = (float*)d_out;                      // [32, 7]

  char* ws = (char*)d_ws;
  float* brG   = (float*)ws;                                   // 32*128*4   = 16384
  float* brnG  = (float*)(ws + 16384);                         // 32*4       = 128
  float* candD = (float*)(ws + 16384 + 128);                   // 32*NBLK*5*4= 250880
  int*   candI = (int*)  (ws + 16384 + 128 + 250880);          // same size
  (void)in_sizes; (void)n_in; (void)out_size; (void)ws_size;

  enc_kernel<<<B_ROWS, 256, 0, stream>>>(x, W_enc, b_enc, brG, brnG);
  dist_topk_kernel<<<NBLK, 256, 0, stream>>>(reps, brG, brnG, candD, candI);
  finalize_kernel<<<1, 256, 0, stream>>>(candD, candI, actions, out);
}
